// PGN_23639499997474
// MI455X (gfx1250) — matrix operand order repaired and checked by reading, not compiled, not run on hardware
//
#include <hip/hip_runtime.h>
#include <math.h>

#define BZ 32
#define LQ 512
#define LC 2048
#define DIM 768
#define VSZ 32000
#define VEXT 34048           // V + LC
#define MQ (BZ * LQ)         // 16384
#define MC (BZ * LC)         // 65536
#define NEG_INF (-__builtin_inff())

typedef float v8f __attribute__((ext_vector_type(8)));
typedef __bf16 v16bf __attribute__((ext_vector_type(16)));
typedef __bf16 bf16_t;

union BF16Frag { v16bf v; bf16_t h[16]; uint4 q[2]; };

__device__ __forceinline__ float sigm(float x) { return 1.f / (1.f + expf(-x)); }

// ---- WMMA fragment loaders -------------------------------------------------
// A 16x32 bf16 layout: lane<16 holds row M=lane, K = {k0+0..7, k0+16..23};
// lanes 16..31 hold same rows with K offset by 8.  off = (lane>>4)*8.
template<bool ADDV>
__device__ __forceinline__ v16bf load_a_f32(const float* rowA, const float* av,
                                            int k0, int off) {
  BF16Frag fr;
#pragma unroll
  for (int i = 0; i < 8; ++i) {
    int ka = k0 + off + i, kb = k0 + 16 + off + i;
    float x0 = rowA[ka], x1 = rowA[kb];
    if (ADDV) { x0 += av[ka]; x1 += av[kb]; }
    fr.h[i]     = (bf16_t)x0;     // native v_cvt_(pk_)bf16_f32
    fr.h[8 + i] = (bf16_t)x1;
  }
  return fr.v;
}

__device__ __forceinline__ v16bf load_a_bf(const bf16_t* rowA, int k0, int off) {
  BF16Frag fr;
  fr.q[0] = *(const uint4*)(rowA + k0 + off);
  fr.q[1] = *(const uint4*)(rowA + k0 + 16 + off);
  return fr.v;
}

// B 32x16 bf16 layout: lane holds column N = lane%16, K = {k0+off+0..7, k0+16+off+0..7}.
__device__ __forceinline__ v16bf load_b16(const bf16_t* p, int ld) {
  BF16Frag fr;
#pragma unroll
  for (int i = 0; i < 8; ++i) {
    fr.h[i]     = p[i * ld];
    fr.h[8 + i] = p[(16 + i) * ld];
  }
  return fr.v;
}

// ---- main GEMM: wave computes 16 rows x 128 cols, K = 768 ------------------
// B tile (32 k x 128 n bf16, 8KB) is staged in LDS per block (double-buffered,
// padded stride 136) so the 8 waves share one set of global B loads.
enum { MODE_PRE = 0, MODE_SC = 1, MODE_AT = 2 };
#define LDSB_STRIDE 136

template<int MODE>
__global__ __launch_bounds__(256) void gemm_k(
    const float* __restrict__ Af32, const bf16_t* __restrict__ Abf,
    const bf16_t* __restrict__ Bt,               // k-major bf16 weight (768x768)
    const float* __restrict__ colbias,           // PRE: aw2[BZ][768]; SC/AT: bias[768]
    const float* __restrict__ addvec,            // AT: hp[BZ][768]
    const float* __restrict__ wvec,              // SC/AT: dot vector [768]
    bf16_t* __restrict__ outbf,                  // PRE: bf16 out [M][768]
    float* __restrict__ outdot,                  // SC/AT: partials [6][M]
    int rows_per_batch, int Mrows)
{
  __shared__ bf16_t ldsB[2][32 * LDSB_STRIDE];

  const int tid = threadIdx.x;
  const int wave = tid >> 5, lane = tid & 31;
  const int lc = lane & 15, half = lane >> 4, off = half * 8;
  const int m0 = blockIdx.x * 128 + wave * 16;
  const int n0 = blockIdx.y * 128;
  const int arow = m0 + lc;

  const float* rowA32 = nullptr;
  const bf16_t* rowAbf = nullptr;
  const float* av = nullptr;
  if (MODE == MODE_SC) {
    rowAbf = Abf + (size_t)arow * DIM;
  } else {
    rowA32 = Af32 + (size_t)arow * DIM;
  }
  if (MODE == MODE_AT) av = addvec + (size_t)(arow / rows_per_batch) * DIM;

  // cooperative B staging: 32 rows x 128 cols, 16 bf16 (32B) per thread
  const int srow = tid >> 3;            // 0..31
  const int schunk = (tid & 7) * 16;    // 0,16,...,112
  const bf16_t* gB = Bt + (size_t)srow * DIM + n0 + schunk;

  v8f acc[8];
#pragma unroll
  for (int t = 0; t < 8; ++t) acc[t] = (v8f){0.f, 0.f, 0.f, 0.f, 0.f, 0.f, 0.f, 0.f};

  // prologue: stage K-step 0
  {
    const uint4* g = (const uint4*)gB;
    uint4* d = (uint4*)&ldsB[0][srow * LDSB_STRIDE + schunk];
    d[0] = g[0];
    d[1] = g[1];
  }
  __syncthreads();

  for (int i = 0; i < DIM / 32; ++i) {
    const int k0 = i * 32;
    // stage next B tile into the other buffer (no hazard: prior reads of that
    // buffer were fenced by the barrier at the end of the previous iteration)
    if (i < DIM / 32 - 1) {
      const uint4* g = (const uint4*)(gB + (size_t)(k0 + 32) * DIM);
      uint4 s0 = g[0], s1 = g[1];
      uint4* d = (uint4*)&ldsB[(i + 1) & 1][srow * LDSB_STRIDE + schunk];
      d[0] = s0;
      d[1] = s1;
    }

    v16bf a = (MODE == MODE_SC) ? load_a_bf(rowAbf, k0, off)
            : (MODE == MODE_AT) ? load_a_f32<true>(rowA32, av, k0, off)
                                : load_a_f32<false>(rowA32, nullptr, k0, off);

    const bf16_t* bbase = &ldsB[i & 1][off * LDSB_STRIDE + lc];
#pragma unroll
    for (int t = 0; t < 8; ++t) {
      v16bf bfr = load_b16(bbase + t * 16, LDSB_STRIDE);
      acc[t] = __builtin_amdgcn_wmma_f32_16x16x32_bf16(
          false, a, false, bfr, (short)0, acc[t], false, false);
    }
    __syncthreads();
  }

  if (MODE == MODE_PRE) {
    // D layout: lanes<16: M=r, N=lane; lanes>=16: M=8+r, N=lane-16
#pragma unroll
    for (int t = 0; t < 8; ++t) {
      int n = n0 + t * 16 + lc;
#pragma unroll
      for (int r = 0; r < 8; ++r) {
        int m = m0 + r + half * 8;
        float v = acc[t][r] + colbias[(size_t)(m / rows_per_batch) * DIM + n];
        outbf[(size_t)m * DIM + n] = (bf16_t)v;
      }
    }
  } else {
    float part[8];
#pragma unroll
    for (int r = 0; r < 8; ++r) part[r] = 0.f;
#pragma unroll
    for (int t = 0; t < 8; ++t) {
      int n = n0 + t * 16 + lc;
      float wv = wvec[n], cb = colbias[n];
#pragma unroll
      for (int r = 0; r < 8; ++r) part[r] += tanhf(acc[t][r] + cb) * wv;
    }
    // reduce over the 16 lanes of each half (columns of this n-group)
#pragma unroll
    for (int s = 1; s < 16; s <<= 1) {
#pragma unroll
      for (int r = 0; r < 8; ++r) part[r] += __shfl_xor(part[r], s, 32);
    }
    if (lc == 0) {
      float* o = outdot + (size_t)blockIdx.y * Mrows + m0 + half * 8;
#pragma unroll
      for (int r = 0; r < 8; ++r) o[r] = part[r];
    }
  }
}

// ---- vocab GEMM: vt[b][v] = hs[b] . W_pgn[v] + b_pgn[v]  (K = 1536) --------
__global__ __launch_bounds__(256) void vt_k(const float* __restrict__ Wpgn,
    const bf16_t* __restrict__ hsT,  // [1536][32] bf16
    const float* __restrict__ bpgn, float* __restrict__ vt)
{
  const int tid = threadIdx.x, wave = tid >> 5, lane = tid & 31;
  const int lc = lane & 15, half = lane >> 4, off = half * 8;
  const int v0 = (blockIdx.x * 8 + wave) * 16;
  const float* rowA = Wpgn + (size_t)(v0 + lc) * (2 * DIM);
  v8f acc[2];
  acc[0] = (v8f){0.f, 0.f, 0.f, 0.f, 0.f, 0.f, 0.f, 0.f};
  acc[1] = acc[0];
  for (int k0 = 0; k0 < 2 * DIM; k0 += 32) {
    v16bf a = load_a_f32<false>(rowA, nullptr, k0, off);
#pragma unroll
    for (int t = 0; t < 2; ++t) {
      v16bf b = load_b16(hsT + (size_t)(k0 + off) * BZ + t * 16 + lc, BZ);
      acc[t] = __builtin_amdgcn_wmma_f32_16x16x32_bf16(
          false, a, false, b, (short)0, acc[t], false, false);
    }
  }
#pragma unroll
  for (int t = 0; t < 2; ++t) {
#pragma unroll
    for (int r = 0; r < 8; ++r) {
      int vv = v0 + r + half * 8;
      int bb = t * 16 + lc;
      vt[(size_t)bb * VSZ + vv] = acc[t][r] + bpgn[vv];
    }
  }
}

// ---- block reductions ------------------------------------------------------
__device__ float blk_max(float v, float* red) {
  int tid = threadIdx.x;
  red[tid] = v; __syncthreads();
  for (int s = 128; s > 0; s >>= 1) {
    if (tid < s) red[tid] = fmaxf(red[tid], red[tid + s]);
    __syncthreads();
  }
  float r = red[0]; __syncthreads();
  return r;
}
__device__ float blk_sum(float v, float* red) {
  int tid = threadIdx.x;
  red[tid] = v; __syncthreads();
  for (int s = 128; s > 0; s >>= 1) {
    if (tid < s) red[tid] += red[tid + s];
    __syncthreads();
  }
  float r = red[0]; __syncthreads();
  return r;
}

// ---- small prep: aw2 = ai@W_att2.T ; h_ = tanh(sum(h)@W_ff1.T + b) ---------
__global__ __launch_bounds__(256) void prep_k(const float* __restrict__ ai,
    const float* __restrict__ Watt2, const float* __restrict__ h,
    const float* __restrict__ Wff1, const float* __restrict__ bff1,
    float* __restrict__ aw2, float* __restrict__ hbuf)
{
  int idx = blockIdx.x * 256 + threadIdx.x;       // 2 * 24576
  int which = idx >= (BZ * DIM);
  int r = idx & (BZ * DIM - 1);
  int b = r & 31, d = r >> 5;                      // wave-uniform d
  if (!which) {
    const float* a = ai + b * DIM;
    const float* w = Watt2 + (size_t)d * DIM;
    float s = 0.f;
    for (int k = 0; k < DIM; ++k) s += a[k] * w[k];
    aw2[b * DIM + d] = s;
  } else {
    const float* h0 = h + b * DIM;
    const float* h1 = h + BZ * DIM + b * DIM;
    const float* w = Wff1 + (size_t)d * DIM;
    float s = bff1[d];
    for (int k = 0; k < DIM; ++k) s += (h0[k] + h1[k]) * w[k];
    hbuf[b * DIM + d] = tanhf(s);
  }
}

// ---- transpose+convert the three 768x768 weights to k-major bf16 -----------
__global__ __launch_bounds__(256) void convw_k(const float* __restrict__ W1,
    const float* __restrict__ W2, const float* __restrict__ W3,
    bf16_t* __restrict__ T1, bf16_t* __restrict__ T2, bf16_t* __restrict__ T3)
{
  int idx = blockIdx.x * 256 + threadIdx.x;       // 3 * 589824
  int w = idx / (DIM * DIM);
  int e = idx - w * (DIM * DIM);
  int k = e / DIM, n = e - k * DIM;
  const float* src = (w == 0) ? W1 : (w == 1) ? W2 : W3;
  bf16_t* dst = (w == 0) ? T1 : (w == 1) ? T2 : T3;
  dst[e] = (bf16_t)src[(size_t)n * DIM + k];
}

__global__ __launch_bounds__(256) void fill_k(float* p, float v, int n) {
  int i = blockIdx.x * 256 + threadIdx.x;
  if (i < n) p[i] = v;
}

// ---- softmax pool over q -> hp = h_ + pooled -------------------------------
__global__ __launch_bounds__(256) void pool_q_k(const float* __restrict__ sc6,
    const float* __restrict__ bfa2, const int* __restrict__ qm,
    const float* __restrict__ q, const float* __restrict__ hbuf,
    float* __restrict__ hp)
{
  __shared__ float red[256];
  __shared__ float wbuf[LQ];
  int b = blockIdx.x, tid = threadIdx.x;
  float bias = bfa2[0];
  float lmax = NEG_INF;
  for (int l = tid; l < LQ; l += 256) {
    float v;
    if (qm[b * LQ + l] == 0) v = NEG_INF;
    else {
      v = bias;
      for (int g = 0; g < 6; ++g) v += sc6[(size_t)g * MQ + b * LQ + l];
    }
    wbuf[l] = v;
    lmax = fmaxf(lmax, v);
  }
  float mx = blk_max(lmax, red);
  float lsum = 0.f;
  for (int l = tid; l < LQ; l += 256) {
    float e = expf(wbuf[l] - mx);
    wbuf[l] = e;
    lsum += e;
  }
  float Z = blk_sum(lsum, red);
  float inv = 1.f / Z;
  for (int l = tid; l < LQ; l += 256) wbuf[l] *= inv;
  __syncthreads();
  for (int d = tid; d < DIM; d += 256) {
    float a = 0.f;
    for (int l = 0; l < LQ; ++l) a += wbuf[l] * q[((size_t)b * LQ + l) * DIM + d];
    hp[b * DIM + d] = hbuf[b * DIM + d] + a;
  }
}

// ---- softmax pool over at -> y ; also stores masked at back ----------------
__global__ __launch_bounds__(256) void pool_at_k(const float* __restrict__ at6,
    const float* __restrict__ bff3b, const int* __restrict__ cm,
    const float* __restrict__ Y, float* __restrict__ at, float* __restrict__ ybuf)
{
  __shared__ float red[256];
  __shared__ float wbuf[LC];
  int b = blockIdx.x, tid = threadIdx.x;
  float bias = bff3b[0];
  float lmax = NEG_INF;
  for (int l = tid; l < LC; l += 256) {
    float v;
    if (cm[b * LC + l] == 0) v = NEG_INF;
    else {
      v = bias;
      for (int g = 0; g < 6; ++g) v += at6[(size_t)g * MC + b * LC + l];
    }
    at[b * LC + l] = v;          // masked pre-softmax scores kept for OOV max
    wbuf[l] = v;
    lmax = fmaxf(lmax, v);
  }
  float mx = blk_max(lmax, red);
  float lsum = 0.f;
  for (int l = tid; l < LC; l += 256) {
    float e = expf(wbuf[l] - mx);
    wbuf[l] = e;
    lsum += e;
  }
  float Z = blk_sum(lsum, red);
  float inv = 1.f / Z;
  for (int l = tid; l < LC; l += 256) wbuf[l] *= inv;
  __syncthreads();
  for (int d = tid; d < DIM; d += 256) {
    float a = 0.f;
    for (int l = 0; l < LC; ++l) a += wbuf[l] * Y[((size_t)b * LC + l) * DIM + d];
    ybuf[b * DIM + d] = a;
  }
}

// ---- LSTM cell (one thread per (b,d); wave-uniform weight rows) ------------
__global__ __launch_bounds__(256) void lstm_k(const float* __restrict__ x0,
    const float* __restrict__ x1, const float* __restrict__ hprev,
    const float* __restrict__ cprev, const float* __restrict__ Wih,
    const float* __restrict__ Whh, const float* __restrict__ bih,
    const float* __restrict__ bhh, float* __restrict__ hout,
    float* __restrict__ cout, bf16_t* __restrict__ hsT,
    int two_inputs, int hoff)
{
  int idx = blockIdx.x * 256 + threadIdx.x;   // 24576
  int b = idx & 31, d = idx >> 5;
  const float* xa = x0 + b * DIM;
  const float* xb = x1 ? x1 + b * DIM : nullptr;
  const float* hh = hprev + b * DIM;
  int xw = two_inputs ? 2 * DIM : DIM;
  float g[4];
#pragma unroll
  for (int gi = 0; gi < 4; ++gi) {
    int r = gi * DIM + d;
    const float* wi = Wih + (size_t)r * xw;
    const float* wh = Whh + (size_t)r * DIM;
    float s = bih[r] + bhh[r];
    for (int k = 0; k < DIM; ++k) s += xa[k] * wi[k];
    if (two_inputs) for (int k = 0; k < DIM; ++k) s += xb[k] * wi[DIM + k];
    for (int k = 0; k < DIM; ++k) s += hh[k] * wh[k];
    g[gi] = s;
  }
  float cn = sigm(g[1]) * cprev[b * DIM + d] + sigm(g[0]) * tanhf(g[2]);
  float hn = sigm(g[3]) * tanhf(cn);
  hout[b * DIM + d] = hn;
  cout[b * DIM + d] = cn;
  hsT[(size_t)(hoff + d) * BZ + b] = (bf16_t)hn;
}

// ---- OOV scatter max / gather ----------------------------------------------
__device__ __forceinline__ void atomicMaxF(float* a, float v) {
  if (v >= 0.f) atomicMax((int*)a, __float_as_int(v));
  else atomicMin((unsigned int*)a, __float_as_uint(v));
}

__global__ __launch_bounds__(256) void oovmax_k(const int* __restrict__ cids,
    const float* __restrict__ at, float* __restrict__ mbuf)
{
  int idx = blockIdx.x * 256 + threadIdx.x;   // 65536
  int b = idx >> 11;
  int id = cids[idx];
  int oov = id >= VSZ;
  int tgt = oov ? (id - VSZ) : 0;
  float val = oov ? at[idx] : 0.f;
  atomicMaxF(&mbuf[b * LC + tgt], val);
}

__global__ __launch_bounds__(256) void oovgather_k(const int* __restrict__ cids,
    const float* __restrict__ mbuf, float* __restrict__ maxo)
{
  int idx = blockIdx.x * 256 + threadIdx.x;
  int b = idx >> 11;
  int id = cids[idx];
  int tgt = (id >= VSZ) ? (id - VSZ) : 0;
  float a0 = mbuf[b * LC + tgt];
  maxo[idx] = (a0 != 0.f) ? a0 : NEG_INF;
}

// ---- extended softmax ------------------------------------------------------
__global__ __launch_bounds__(256) void smred_k(const float* __restrict__ vt,
    const float* __restrict__ maxo, float* __restrict__ mxZ)
{
  __shared__ float red[256];
  int b = blockIdx.x, tid = threadIdx.x;
  float lmax = NEG_INF;
  for (int i = tid; i < VEXT; i += 256) {
    float v = (i < VSZ) ? vt[(size_t)b * VSZ + i] : maxo[b * LC + (i - VSZ)];
    lmax = fmaxf(lmax, v);
  }
  float mx = blk_max(lmax, red);
  float ls = 0.f;
  for (int i = tid; i < VEXT; i += 256) {
    float v = (i < VSZ) ? vt[(size_t)b * VSZ + i] : maxo[b * LC + (i - VSZ)];
    ls += expf(v - mx);
  }
  float Z = blk_sum(ls, red);
  if (tid == 0) { mxZ[b * 2] = mx; mxZ[b * 2 + 1] = Z; }
}

__global__ __launch_bounds__(256) void smwrite_k(const float* __restrict__ vt,
    const float* __restrict__ mxZ, float* __restrict__ wt)
{
  int idx = blockIdx.x * 256 + threadIdx.x;   // 32 * 34048
  int b = idx / VEXT, i = idx - b * VEXT;
  float out = 0.f;
  if (i < VSZ) out = expf(vt[(size_t)b * VSZ + i] - mxZ[b * 2]) / mxZ[b * 2 + 1];
  wt[idx] = out;
}

__global__ __launch_bounds__(256) void smscat_k(const int* __restrict__ cids,
    const float* __restrict__ maxo, const float* __restrict__ mxZ,
    float* __restrict__ wt)
{
  int idx = blockIdx.x * 256 + threadIdx.x;   // 65536
  int b = idx >> 11;
  float mo = maxo[idx];
  if (mo > -1e30f) {
    float p = expf(mo - mxZ[b * 2]) / mxZ[b * 2 + 1];
    atomicAdd(&wt[(size_t)b * VEXT + cids[idx]], p);
  }
}

// ---------------------------------------------------------------------------
extern "C" void kernel_launch(void* const* d_in, const int* in_sizes, int n_in,
                              void* d_out, int out_size, void* d_ws, size_t ws_size,
                              hipStream_t stream) {
  (void)in_sizes; (void)n_in; (void)out_size; (void)ws_size;
  const float* Y     = (const float*)d_in[0];
  const float* h     = (const float*)d_in[1];
  const float* c     = (const float*)d_in[2];
  const float* ai    = (const float*)d_in[3];
  const float* q     = (const float*)d_in[4];
  const int*   qm    = (const int*)d_in[5];
  const int*   cids  = (const int*)d_in[6];
  const int*   cm    = (const int*)d_in[7];
  const float* Watt1 = (const float*)d_in[8];
  const float* Watt2 = (const float*)d_in[9];
  const float* Wfa1  = (const float*)d_in[10];
  const float* bfa1  = (const float*)d_in[11];
  const float* wfa2  = (const float*)d_in[12];
  const float* bfa2  = (const float*)d_in[13];
  const float* Wff1  = (const float*)d_in[14];
  const float* bff1  = (const float*)d_in[15];
  const float* Wff3a = (const float*)d_in[16];
  const float* bff3a = (const float*)d_in[17];
  const float* wff3b = (const float*)d_in[18];
  const float* bff3b = (const float*)d_in[19];
  const float* Wih0  = (const float*)d_in[20];
  const float* Whh0  = (const float*)d_in[21];
  const float* bih0  = (const float*)d_in[22];
  const float* bhh0  = (const float*)d_in[23];
  const float* Wih1  = (const float*)d_in[24];
  const float* Whh1  = (const float*)d_in[25];
  const float* bih1  = (const float*)d_in[26];
  const float* bhh1  = (const float*)d_in[27];
  const float* Wpgn  = (const float*)d_in[28];
  const float* bpgn  = (const float*)d_in[29];

  float* wt   = (float*)d_out;                  // [32][34048]
  float* hnew = wt + (size_t)BZ * VEXT;         // [2][32][768]
  float* cnew = hnew + 2 * BZ * DIM;            // [2][32][768]

  char* ws = (char*)d_ws;
  size_t off = 0;
  auto alloc = [&](size_t bytes) -> void* {
    void* p = ws + off;
    off = (off + bytes + 255) & ~(size_t)255;
    return p;
  };
  bf16_t* WT1   = (bf16_t*)alloc((size_t)DIM * DIM * 2);
  bf16_t* WTF   = (bf16_t*)alloc((size_t)DIM * DIM * 2);
  bf16_t* WTA   = (bf16_t*)alloc((size_t)DIM * DIM * 2);
  bf16_t* PREBF = (bf16_t*)alloc((size_t)MQ * DIM * 2);
  float* AW2  = (float*)alloc((size_t)BZ * DIM * 4);
  float* HBUF = (float*)alloc((size_t)BZ * DIM * 4);
  float* SC6  = (float*)alloc((size_t)6 * MQ * 4);
  float* AT6  = (float*)alloc((size_t)6 * MC * 4);
  float* ATB  = (float*)alloc((size_t)MC * 4);
  float* HP   = (float*)alloc((size_t)BZ * DIM * 4);
  float* YB   = (float*)alloc((size_t)BZ * DIM * 4);
  bf16_t* HST = (bf16_t*)alloc((size_t)2 * DIM * BZ * 2);
  float* VT   = (float*)alloc((size_t)BZ * VSZ * 4);
  float* MBUF = (float*)alloc((size_t)MC * 4);
  float* MAXO = (float*)alloc((size_t)MC * 4);
  float* MXZ  = (float*)alloc((size_t)BZ * 2 * 4);

  // 1. init OOV max buffer to -inf
  fill_k<<<MC / 256, 256, 0, stream>>>(MBUF, NEG_INF, MC);
  // 2. small GEMMs: aw2, h_
  prep_k<<<(2 * BZ * DIM) / 256, 256, 0, stream>>>(ai, Watt2, h, Wff1, bff1, AW2, HBUF);
  // 3. weights -> k-major bf16
  convw_k<<<(3 * DIM * DIM) / 256, 256, 0, stream>>>(Watt1, Wfa1, Wff3a, WT1, WTF, WTA);
  // 4. pre = q @ W_att1.T + aw2   (bf16 out)
  gemm_k<MODE_PRE><<<dim3(MQ / 128, 6), 256, 0, stream>>>(
      q, nullptr, WT1, AW2, nullptr, nullptr, PREBF, nullptr, LQ, MQ);
  // 5. sc partials = tanh(pre @ W_fa1.T + b_fa1) . w_fa2
  gemm_k<MODE_SC><<<dim3(MQ / 128, 6), 256, 0, stream>>>(
      nullptr, PREBF, WTF, bfa1, nullptr, wfa2, nullptr, SC6, LQ, MQ);
  // 6. softmax-pool q -> hp = h_ + pooled
  pool_q_k<<<BZ, 256, 0, stream>>>(SC6, bfa2, qm, q, HBUF, HP);
  // 7. at partials = tanh((Y + hp) @ W_ff3a.T + b_ff3a) . w_ff3b
  gemm_k<MODE_AT><<<dim3(MC / 128, 6), 256, 0, stream>>>(
      Y, nullptr, WTA, bff3a, HP, wff3b, nullptr, AT6, LC, MC);
  // 8. softmax-pool Y -> ybuf; masked at -> ATB
  pool_at_k<<<BZ, 256, 0, stream>>>(AT6, bff3b, cm, Y, ATB, YB);
  // 9/10. LSTM layers
  lstm_k<<<(BZ * DIM) / 256, 256, 0, stream>>>(
      YB, ai, h, c, Wih0, Whh0, bih0, bhh0,
      hnew, cnew, HST, 1, 0);
  lstm_k<<<(BZ * DIM) / 256, 256, 0, stream>>>(
      hnew, nullptr, h + BZ * DIM, c + BZ * DIM, Wih1, Whh1, bih1, bhh1,
      hnew + BZ * DIM, cnew + BZ * DIM, HST, 0, DIM);
  // 11. vocab logits (WMMA, N = 32 batches)
  vt_k<<<VSZ / (8 * 16), 256, 0, stream>>>(Wpgn, HST, bpgn, VT);
  // 12/13. OOV scatter-max + gather
  oovmax_k<<<MC / 256, 256, 0, stream>>>(cids, ATB, MBUF);
  oovgather_k<<<MC / 256, 256, 0, stream>>>(cids, MBUF, MAXO);
  // 14-16. extended softmax + scatter-add
  smred_k<<<BZ, 256, 0, stream>>>(VT, MAXO, MXZ);
  smwrite_k<<<(BZ * VEXT) / 256, 256, 0, stream>>>(VT, MXZ, wt);
  smscat_k<<<MC / 256, 256, 0, stream>>>(cids, MAXO, MXZ, wt);
}
